// GameOutcomePredictorV4_33784212750365
// MI455X (gfx1250) — matrix operand order repaired and checked by reading, not compiled, not run on hardware
//
#include <hip/hip_runtime.h>
#include <hip/hip_bf16.h>

typedef __attribute__((ext_vector_type(16))) _Float16 v16h;
typedef __attribute__((ext_vector_type(8)))  _Float16 v8h;
typedef __attribute__((ext_vector_type(8)))  float    v8f;

#define C_IN 22
#define HDIM 64
#define TCH  32
#define KW   3
#define NCLS 3
#define BSZ  16
#define BN_EPS 1e-5f

// ---------------------------------------------------------------------------
// Generic fill
// ---------------------------------------------------------------------------
__global__ void fill_kernel(float* __restrict__ p, long long n, float v) {
    long long i = (long long)blockIdx.x * blockDim.x + threadIdx.x;
    long long stride = (long long)gridDim.x * blockDim.x;
    for (; i < n; i += stride) p[i] = v;
}

// ---------------------------------------------------------------------------
// Degree: deg pre-filled with 1.0 (self loops); add edge weights at col.
// ---------------------------------------------------------------------------
__global__ void deg_accum_kernel(const int* __restrict__ ei, const float* __restrict__ ew,
                                 float* __restrict__ deg, int E) {
    int e = blockIdx.x * blockDim.x + threadIdx.x;
    if (e >= E) return;
    atomicAdd(&deg[ei[E + e]], ew[e]);
}

__global__ void rsqrt_inplace_kernel(float* __restrict__ deg, int N) {
    int i = blockIdx.x * blockDim.x + threadIdx.x;
    if (i >= N) return;
    float d = deg[i];
    deg[i] = (d > 0.f) ? rsqrtf(d) : 0.f;
}

// ---------------------------------------------------------------------------
// Pack A [M,Kact] f32 row-major -> Ah [M,KPAD] f16, zero-padded K tail.
// Pure streaming pass; trivial vs 23.3 TB/s HBM.
// ---------------------------------------------------------------------------
template<int KPAD>
__global__ void pack_a_f16_kernel(const float* __restrict__ A, int Kact,
                                  _Float16* __restrict__ Ah, long long M) {
    long long i = (long long)blockIdx.x * blockDim.x + threadIdx.x;
    long long n = M * KPAD;
    long long stride = (long long)gridDim.x * blockDim.x;
    for (; i < n; i += stride) {
        int k = (int)(i % KPAD);
        long long row = i / KPAD;
        Ah[i] = (k < Kact) ? (_Float16)A[row * Kact + k] : (_Float16)0.f;
    }
}

// Pack B [Kact,64] f32 -> Bt [64,KPAD] f16 transposed, zero-padded. One block.
template<int KPAD>
__global__ void pack_bt_f16_kernel(const float* __restrict__ B, int Kact,
                                   _Float16* __restrict__ Bt) {
    for (int i = threadIdx.x; i < HDIM * KPAD; i += blockDim.x) {
        int n = i / KPAD, k = i % KPAD;
        Bt[i] = (k < Kact) ? (_Float16)B[(size_t)k * HDIM + n] : (_Float16)0.f;
    }
}

// ---------------------------------------------------------------------------
// WMMA GEMM: C[M,64] = Ah[M,KPAD] * Bt[64,KPAD]^T  (both f16, pre-packed).
// Block = 128 threads = 4 waves; wave w owns columns n0=16w; block owns 16
// rows. Fragments built from unconditional b128 loads (no predication):
//   A frag (16x32 f16, ISA layout): lane (half,m) holds K-locals
//     {half*8+j} and {16+half*8+j}  -> two contiguous 8xf16 chunks.
//   B frag (32x16 f16): lane (half,n) holds K-locals
//     {half*8+j} and {16+half*8+j}  -> two contiguous 8xf16 chunks.
// ---------------------------------------------------------------------------
template<int KPAD>
__global__ void gemm_wmma_n64_kernel(const _Float16* __restrict__ Ah,
                                     const _Float16* __restrict__ Bt,
                                     float* __restrict__ C) {
    const int wave = threadIdx.x >> 5;
    const int lane = threadIdx.x & 31;
    const int half = lane >> 4;       // 0: lanes 0-15, 1: lanes 16-31
    const int ml   = lane & 15;       // M (for A/D) or N (for B/D) sub-index
    const int row0 = blockIdx.x * 16;
    const int n0   = wave * 16;

    const _Float16* arow = Ah + (size_t)(row0 + ml) * KPAD;
    const _Float16* brow = Bt + (size_t)(n0 + ml) * KPAD;

    v8f acc = {};
    #pragma unroll
    for (int k0 = 0; k0 < KPAD; k0 += 32) {
        v8h a_lo = *(const v8h*)(arow + k0 + half * 8);        // K-locals half*8 .. +7
        v8h a_hi = *(const v8h*)(arow + k0 + 16 + half * 8);   // K-locals 16+half*8 .. +7
        v8h b_lo = *(const v8h*)(brow + k0 + half * 8);        // K-locals half*8 .. +7
        v8h b_hi = *(const v8h*)(brow + k0 + 16 + half * 8);   // K-locals 16+half*8 .. +7
        v16h af = __builtin_shufflevector(a_lo, a_hi, 0,1,2,3,4,5,6,7,8,9,10,11,12,13,14,15);
        v16h bf = __builtin_shufflevector(b_lo, b_hi, 0,1,2,3,4,5,6,7,8,9,10,11,12,13,14,15);
        acc = __builtin_amdgcn_wmma_f32_16x16x32_f16(
            /*neg_a=*/false, af, /*neg_b=*/false, bf,
            /*c_mod=*/(short)0, acc, /*reuse_a=*/false, /*reuse_b=*/false);
    }
    // D: VGPR r, lane (half,ml) -> element (m = r + 8*half, n = ml)
    #pragma unroll
    for (int r = 0; r < 8; ++r)
        C[(size_t)(row0 + r + 8 * half) * HDIM + n0 + ml] = acc[r];
}

// ---------------------------------------------------------------------------
// Sparse aggregation: out[col] += dinv[row]*w*dinv[col] * h[row].
// Edges e < E come from edge_index/edge_weight; e >= E are self loops (w=1).
// 4 threads per edge, 16 features each (4x float4 gathers, 16 f32 atomics).
// h (16MB) and out (16MB) are L2-resident on MI455X (192MB L2), so this is
// L2-atomic-throughput bound, not HBM bound.
// ---------------------------------------------------------------------------
__global__ void aggregate_kernel(const float* __restrict__ h, const int* __restrict__ ei,
                                 const float* __restrict__ ew, const float* __restrict__ dinv,
                                 float* __restrict__ out, int E, int N) {
    long long tid   = (long long)blockIdx.x * blockDim.x + threadIdx.x;
    long long total = (long long)(E + N) * 4;
    if (tid >= total) return;
    int e    = (int)(tid >> 2);
    int part = (int)(tid & 3);
    int r, c; float w;
    if (e < E) { r = ei[e]; c = ei[E + e]; w = ew[e]; }
    else       { r = c = e - E; w = 1.0f; }
    float nrm = dinv[r] * w * dinv[c];
    const float4* src = (const float4*)(h + (size_t)r * HDIM + part * 16);
    float* dst = out + (size_t)c * HDIM + part * 16;
    #pragma unroll
    for (int q = 0; q < 4; ++q) {
        float4 v = src[q];
        atomicAdd(dst + 4 * q + 0, nrm * v.x);
        atomicAdd(dst + 4 * q + 1, nrm * v.y);
        atomicAdd(dst + 4 * q + 2, nrm * v.z);
        atomicAdd(dst + 4 * q + 3, nrm * v.w);
    }
}

// ---------------------------------------------------------------------------
// y = relu(y + bias[c]) over [N, 64]
// ---------------------------------------------------------------------------
__global__ void bias_relu_kernel(float* __restrict__ y, const float* __restrict__ b, long long n) {
    long long i = (long long)blockIdx.x * blockDim.x + threadIdx.x;
    long long stride = (long long)gridDim.x * blockDim.x;
    for (; i < n; i += stride) {
        float v = y[i] + b[i & (HDIM - 1)];
        y[i] = fmaxf(v, 0.f);
    }
}

// ---------------------------------------------------------------------------
// Conv1: x viewed as [B, L, 64] node-major (h2 layout). One block per (b,t).
// y[b,t,l] = bias[t] + sum_{j,c} x[b, l+j-2, c] * w[t,c,j], l in [0, L+2).
// Also accumulates BN sums per channel t.
// ---------------------------------------------------------------------------
__global__ void conv1_kernel(const float* __restrict__ x, const float* __restrict__ w,
                             const float* __restrict__ bias, float* __restrict__ y,
                             float* __restrict__ sum, float* __restrict__ sumsq, int L) {
    const int b = blockIdx.x / TCH;
    const int t = blockIdx.x % TCH;
    __shared__ float wsh[HDIM * KW];
    for (int i = threadIdx.x; i < HDIM * KW; i += blockDim.x)
        wsh[i] = w[(size_t)t * HDIM * KW + i];
    __syncthreads();

    const int Lout = L + 2;
    const float bv = bias[t];
    float bsum = 0.f, bsq = 0.f;
    for (int l = threadIdx.x; l < Lout; l += blockDim.x) {
        float acc = bv;
        #pragma unroll
        for (int j = 0; j < KW; ++j) {
            int xi = l + j - 2;
            if (xi >= 0 && xi < L) {
                const float* xp = x + ((size_t)b * L + xi) * HDIM;
                #pragma unroll
                for (int c = 0; c < HDIM; c += 4) {
                    float4 xv = *(const float4*)(xp + c);
                    acc = fmaf(xv.x, wsh[(c + 0) * KW + j], acc);
                    acc = fmaf(xv.y, wsh[(c + 1) * KW + j], acc);
                    acc = fmaf(xv.z, wsh[(c + 2) * KW + j], acc);
                    acc = fmaf(xv.w, wsh[(c + 3) * KW + j], acc);
                }
            }
        }
        y[((size_t)(b * TCH + t)) * Lout + l] = acc;
        bsum += acc; bsq += acc * acc;
    }
    __shared__ float rs[256], rq[256];
    rs[threadIdx.x] = bsum; rq[threadIdx.x] = bsq;
    __syncthreads();
    for (int s = 128; s > 0; s >>= 1) {
        if (threadIdx.x < s) { rs[threadIdx.x] += rs[threadIdx.x + s]; rq[threadIdx.x] += rq[threadIdx.x + s]; }
        __syncthreads();
    }
    if (threadIdx.x == 0) { atomicAdd(&sum[t], rs[0]); atomicAdd(&sumsq[t], rq[0]); }
}

// ---------------------------------------------------------------------------
// BN apply + ReLU in place over [B, TCH, Lout]
// ---------------------------------------------------------------------------
__global__ void bn_relu_kernel(float* __restrict__ y, const float* __restrict__ sum,
                               const float* __restrict__ sumsq, const float* __restrict__ g,
                               const float* __restrict__ beta, int Lout, float invCount) {
    long long n = (long long)BSZ * TCH * Lout;
    long long i = (long long)blockIdx.x * blockDim.x + threadIdx.x;
    long long stride = (long long)gridDim.x * blockDim.x;
    for (; i < n; i += stride) {
        int t = (int)((i / Lout) % TCH);
        float m  = sum[t] * invCount;
        float vv = sumsq[t] * invCount - m * m;
        float v = (y[i] - m) * rsqrtf(vv + BN_EPS) * g[t] + beta[t];
        y[i] = fmaxf(v, 0.f);
    }
}

// ---------------------------------------------------------------------------
// Conv2: x2 [B, TCH, Lin] channel-major. One block per (b,t).
// ---------------------------------------------------------------------------
__global__ void conv2_kernel(const float* __restrict__ x, const float* __restrict__ w,
                             const float* __restrict__ bias, float* __restrict__ y,
                             float* __restrict__ sum, float* __restrict__ sumsq, int Lin) {
    const int b = blockIdx.x / TCH;
    const int t = blockIdx.x % TCH;
    __shared__ float wsh[TCH * KW];
    for (int i = threadIdx.x; i < TCH * KW; i += blockDim.x)
        wsh[i] = w[(size_t)t * TCH * KW + i];
    __syncthreads();

    const int Lout = Lin + 2;
    const float bv = bias[t];
    float bsum = 0.f, bsq = 0.f;
    for (int l = threadIdx.x; l < Lout; l += blockDim.x) {
        float acc = bv;
        #pragma unroll
        for (int j = 0; j < KW; ++j) {
            int xi = l + j - 2;
            if (xi >= 0 && xi < Lin) {
                #pragma unroll
                for (int c = 0; c < TCH; ++c)
                    acc = fmaf(x[((size_t)(b * TCH + c)) * Lin + xi], wsh[c * KW + j], acc);
            }
        }
        y[((size_t)(b * TCH + t)) * Lout + l] = acc;
        bsum += acc; bsq += acc * acc;
    }
    __shared__ float rs[256], rq[256];
    rs[threadIdx.x] = bsum; rq[threadIdx.x] = bsq;
    __syncthreads();
    for (int s = 128; s > 0; s >>= 1) {
        if (threadIdx.x < s) { rs[threadIdx.x] += rs[threadIdx.x + s]; rq[threadIdx.x] += rq[threadIdx.x + s]; }
        __syncthreads();
    }
    if (threadIdx.x == 0) { atomicAdd(&sum[t], rs[0]); atomicAdd(&sumsq[t], rq[0]); }
}

// ---------------------------------------------------------------------------
// BN2 apply + ReLU + mean over length, fused. One block per (b,t).
// pool[b*TCH + t] = mean_l relu(bn(y2[b,t,l]))
// ---------------------------------------------------------------------------
__global__ void bn2_pool_kernel(const float* __restrict__ y, const float* __restrict__ sum,
                                const float* __restrict__ sumsq, const float* __restrict__ g,
                                const float* __restrict__ beta, float* __restrict__ pool,
                                int Lout, float invCount) {
    const int b = blockIdx.x / TCH;
    const int t = blockIdx.x % TCH;
    const float m  = sum[t] * invCount;
    const float vv = sumsq[t] * invCount - m * m;
    const float sc = rsqrtf(vv + BN_EPS) * g[t];
    const float bt = beta[t];
    float acc = 0.f;
    for (int l = threadIdx.x; l < Lout; l += blockDim.x) {
        float v = (y[((size_t)(b * TCH + t)) * Lout + l] - m) * sc + bt;
        acc += fmaxf(v, 0.f);
    }
    __shared__ float rs[256];
    rs[threadIdx.x] = acc;
    __syncthreads();
    for (int s = 128; s > 0; s >>= 1) {
        if (threadIdx.x < s) rs[threadIdx.x] += rs[threadIdx.x + s];
        __syncthreads();
    }
    if (threadIdx.x == 0) pool[b * TCH + t] = rs[0] / (float)Lout;
}

// ---------------------------------------------------------------------------
// Head: out = relu(pool @ fc1 + b1) @ fc2 + b2. Tiny: one block.
// ---------------------------------------------------------------------------
__global__ void head_kernel(const float* __restrict__ pool,
                            const float* __restrict__ fc1_w, const float* __restrict__ fc1_b,
                            const float* __restrict__ fc2_w, const float* __restrict__ fc2_b,
                            float* __restrict__ out) {
    __shared__ float z[BSZ * HDIM];
    for (int idx = threadIdx.x; idx < BSZ * HDIM; idx += blockDim.x) {
        int b = idx / HDIM, i = idx % HDIM;
        float s = fc1_b[i];
        #pragma unroll
        for (int t = 0; t < TCH; ++t)
            s = fmaf(pool[b * TCH + t], fc1_w[t * HDIM + i], s);
        z[idx] = fmaxf(s, 0.f);
    }
    __syncthreads();
    for (int idx = threadIdx.x; idx < BSZ * NCLS; idx += blockDim.x) {
        int b = idx / NCLS, c = idx % NCLS;
        float s = fc2_b[c];
        #pragma unroll
        for (int i = 0; i < HDIM; ++i)
            s = fmaf(z[b * HDIM + i], fc2_w[i * NCLS + c], s);
        out[idx] = s;
    }
}

// ---------------------------------------------------------------------------
extern "C" void kernel_launch(void* const* d_in, const int* in_sizes, int n_in,
                              void* d_out, int out_size, void* d_ws, size_t ws_size,
                              hipStream_t stream) {
    const float* x     = (const float*)d_in[0];
    const float* ew    = (const float*)d_in[1];
    const float* W1    = (const float*)d_in[2];
    const float* b1    = (const float*)d_in[3];
    const float* W2    = (const float*)d_in[4];
    const float* b2    = (const float*)d_in[5];
    const float* tc1_w = (const float*)d_in[6];
    const float* tc1_b = (const float*)d_in[7];
    const float* bn1_g = (const float*)d_in[8];
    const float* bn1_b = (const float*)d_in[9];
    const float* tc2_w = (const float*)d_in[10];
    const float* tc2_b = (const float*)d_in[11];
    const float* bn2_g = (const float*)d_in[12];
    const float* bn2_b = (const float*)d_in[13];
    const float* fc1_w = (const float*)d_in[14];
    const float* fc1_b = (const float*)d_in[15];
    const float* fc2_w = (const float*)d_in[16];
    const float* fc2_b = (const float*)d_in[17];
    const int*   ei    = (const int*)d_in[18];
    float*       out   = (float*)d_out;

    const int N = in_sizes[0] / C_IN;   // 65536
    const int E = in_sizes[1];          // 1048576
    const int L = N / BSZ;              // 4096
    const int L1 = L + 2;               // conv1 out length
    const int L2 = L1 + 2;              // conv2 out length

    // Workspace layout (floats, then f16 pack area)
    float* W = (float*)d_ws;
    float* deg   = W;                                    // N (becomes dinv)
    float* bufA  = deg  + N;                             // N*64
    float* bufB  = bufA + (size_t)N * HDIM;              // N*64
    float* sum1  = bufB + (size_t)N * HDIM;              // 32
    float* sq1   = sum1 + TCH;                           // 32
    float* sum2  = sq1  + TCH;                           // 32
    float* sq2   = sum2 + TCH;                           // 32
    float* pool  = sq2  + TCH;                           // BSZ*TCH
    _Float16* packA = (_Float16*)(pool + BSZ * TCH);     // N*64 f16 (max KPAD)
    _Float16* packB = packA + (size_t)N * HDIM;          // 64*64 f16

    const int T256 = 256;
    const long long nh = (long long)N * HDIM;

    // --- degree / normalization ---
    fill_kernel<<<256, T256, 0, stream>>>(deg, N, 1.0f);                       // self-loop weight
    deg_accum_kernel<<<(E + T256 - 1) / T256, T256, 0, stream>>>(ei, ew, deg, E);
    rsqrt_inplace_kernel<<<(N + T256 - 1) / T256, T256, 0, stream>>>(deg, N);

    // --- GCN layer 1: h_pre = x @ W1 (K = 22 padded to 32) ---
    pack_a_f16_kernel<32><<<2048, T256, 0, stream>>>(x, C_IN, packA, N);
    pack_bt_f16_kernel<32><<<1, T256, 0, stream>>>(W1, C_IN, packB);
    gemm_wmma_n64_kernel<32><<<N / 16, 128, 0, stream>>>(packA, packB, bufA);
    fill_kernel<<<2048, T256, 0, stream>>>(bufB, nh, 0.0f);
    {
        long long tw = (long long)(E + N) * 4;
        aggregate_kernel<<<(int)((tw + T256 - 1) / T256), T256, 0, stream>>>(bufA, ei, ew, deg, bufB, E, N);
    }
    bias_relu_kernel<<<2048, T256, 0, stream>>>(bufB, b1, nh);                 // h1 in bufB

    // --- GCN layer 2: h2_pre = h1 @ W2 (K = 64) ---
    pack_a_f16_kernel<64><<<2048, T256, 0, stream>>>(bufB, HDIM, packA, N);
    pack_bt_f16_kernel<64><<<1, T256, 0, stream>>>(W2, HDIM, packB);
    gemm_wmma_n64_kernel<64><<<N / 16, 128, 0, stream>>>(packA, packB, bufA);
    fill_kernel<<<2048, T256, 0, stream>>>(bufB, nh, 0.0f);
    {
        long long tw = (long long)(E + N) * 4;
        aggregate_kernel<<<(int)((tw + T256 - 1) / T256), T256, 0, stream>>>(bufA, ei, ew, deg, bufB, E, N);
    }
    bias_relu_kernel<<<2048, T256, 0, stream>>>(bufB, b2, nh);                 // h2 in bufB

    // --- temporal convs ---
    fill_kernel<<<1, T256, 0, stream>>>(sum1, 4 * TCH + BSZ * TCH, 0.0f);      // stats + pool
    conv1_kernel<<<BSZ * TCH, T256, 0, stream>>>(bufB, tc1_w, tc1_b, bufA, sum1, sq1, L);
    bn_relu_kernel<<<2048, T256, 0, stream>>>(bufA, sum1, sq1, bn1_g, bn1_b, L1,
                                              1.0f / (float)(BSZ * L1));
    conv2_kernel<<<BSZ * TCH, T256, 0, stream>>>(bufA, tc2_w, tc2_b, bufB, sum2, sq2, L1);
    bn2_pool_kernel<<<BSZ * TCH, T256, 0, stream>>>(bufB, sum2, sq2, bn2_g, bn2_b, pool, L2,
                                                    1.0f / (float)(BSZ * L2));

    // --- head ---
    head_kernel<<<1, 256, 0, stream>>>(pool, fc1_w, fc1_b, fc2_w, fc2_b, out);
}
